// Head_40604620817011
// MI455X (gfx1250) — matrix operand order repaired and checked by reading, not compiled, not run on hardware
//
#include <hip/hip_runtime.h>

// ---------------------------------------------------------------------------
// MI455X (gfx1250) fused attention:
//   kernel 0: Wq/Wk/Wv f32 [C,H]  ->  Wt f16 [3][H,C]   (transpose + convert)
//   kernel 1: Q = x@Wq, K = y@Wk, V^T = (y@Wv)^T        (f32 in -> f16 out)
//   kernel 2: flash attention (causal, online softmax)  (f16 WMMA -> f32 out)
// HBM-bound (~285 MB vs 17 GFLOP at 23.3 TB/s): intermediates kept f16 in ws,
// every WMMA fragment is a single aligned vector load, and both hot loops are
// software-pipelined so loads for step i+1 issue before the WMMAs of step i.
// ---------------------------------------------------------------------------

typedef __attribute__((ext_vector_type(16))) _Float16 v16h;
typedef __attribute__((ext_vector_type(8)))  _Float16 v8h;
typedef __attribute__((ext_vector_type(8)))  float    v8f;
typedef __attribute__((ext_vector_type(4)))  float    f4;

#define WMMA(a, b, c) \
  __builtin_amdgcn_wmma_f32_16x16x32_f16(false, (a), false, (b), (short)0, (c), false, false)

static __device__ __forceinline__ v16h cat16(v8h lo, v8h hi) {
  return __builtin_shufflevector(lo, hi, 0,1,2,3,4,5,6,7,8,9,10,11,12,13,14,15);
}

#define NB 256   // batch
#define NT 256   // sequence length
#define NC 512   // channels
#define NH 64    // head size

// ---------------------------------------------------------------------------
// Kernel 0: weight prep.  Wt[proj][col][k] = (f16) W[k][col].
// ---------------------------------------------------------------------------
__global__ __launch_bounds__(256) void attn_wprep_kernel(
    const float* __restrict__ Wq, const float* __restrict__ Wk,
    const float* __restrict__ Wv, _Float16* __restrict__ Wt)
{
  const int flat = blockIdx.x * 256 + threadIdx.x;   // 0 .. 98303
  const int proj = flat >> 15;                       // / (NH*NC)
  const int rem  = flat & 32767;
  const int k    = rem >> 6;
  const int col  = rem & 63;
  const float* W = (proj == 0) ? Wq : ((proj == 1) ? Wk : Wv);
  Wt[proj * (NH * NC) + col * NC + k] = (_Float16)W[k * NH + col];
}

// ---------------------------------------------------------------------------
// Kernel 1: projections. grid = (M/128, 3), block = 256 (8 waves).
// Wave computes a 16x64 tile; software-pipelined K-loop over C in 32-chunks.
// ---------------------------------------------------------------------------
__global__ __launch_bounds__(256) void attn_proj_kernel(
    const float* __restrict__ x, const float* __restrict__ y,
    const _Float16* __restrict__ Wt,
    _Float16* __restrict__ Qh, _Float16* __restrict__ Kh,
    _Float16* __restrict__ Vt)
{
  const int proj = blockIdx.y;
  const int wave = threadIdx.x >> 5;
  const int lane = threadIdx.x & 31;
  const int l15  = lane & 15;
  const int hi   = lane >> 4;            // which 16-lane half
  const int r0   = blockIdx.x * 128 + wave * 16;

  const float*    src = (proj == 0) ? x : y;
  const _Float16* Wp  = Wt + (size_t)proj * (NH * NC);

  const int arow  = r0 + l15;            // A-frag: row = lane&15
  const int abase = hi ? 16 : 0;         // A-frag K-half split
  const int kb    = hi ? 16 : 0;         // B-frag K-half split

  const float*    abp = src + (size_t)arow * NC + abase;  // A base
  const _Float16* wbp = Wp + kb;                           // B base

  // A fragment loader: 16 rows x 32 K of src, f32 -> f16 (4 x b128 / lane)
  auto loadA = [&](int kk) -> v16h {
    f4 a0 = *(const f4*)(abp + kk);
    f4 a1 = *(const f4*)(abp + kk + 4);
    f4 a2 = *(const f4*)(abp + kk + 8);
    f4 a3 = *(const f4*)(abp + kk + 12);
    v16h a;
#pragma unroll
    for (int i = 0; i < 4; ++i) {
      a[i]      = (_Float16)a0[i];
      a[4 + i]  = (_Float16)a1[i];
      a[8 + i]  = (_Float16)a2[i];
      a[12 + i] = (_Float16)a3[i];
    }
    return a;
  };
  // B fragment loader: one 32B load (16 contiguous halfs of a Wt row)
  auto loadB = [&](int nt, int kk) -> v16h {
    return *(const v16h*)(wbp + (size_t)(nt * 16 + l15) * NC + kk);
  };

  v8f acc[4] = {};                       // 16x64 f32 accumulators (4 N-tiles)

  // ---- software pipeline: preload step kk while computing step kk-32
  v16h a = loadA(0);
  v16h b0 = loadB(0, 0), b1 = loadB(1, 0), b2 = loadB(2, 0), b3 = loadB(3, 0);
  for (int kk = 32; kk < NC; kk += 32) {
    v16h an  = loadA(kk);
    v16h bn0 = loadB(0, kk), bn1 = loadB(1, kk);
    v16h bn2 = loadB(2, kk), bn3 = loadB(3, kk);
    acc[0] = WMMA(a, b0, acc[0]);
    acc[1] = WMMA(a, b1, acc[1]);
    acc[2] = WMMA(a, b2, acc[2]);
    acc[3] = WMMA(a, b3, acc[3]);
    a = an; b0 = bn0; b1 = bn1; b2 = bn2; b3 = bn3;
  }
  acc[0] = WMMA(a, b0, acc[0]);
  acc[1] = WMMA(a, b1, acc[1]);
  acc[2] = WMMA(a, b2, acc[2]);
  acc[3] = WMMA(a, b3, acc[3]);

  // ---- store: C/D layout  M = r + 8*hi, N = lane&15
  if (proj < 2) {
    _Float16* dst = (proj == 0) ? Qh : Kh;
#pragma unroll
    for (int nt = 0; nt < 4; ++nt)
#pragma unroll
      for (int r = 0; r < 8; ++r) {
        const int row = r0 + r + 8 * hi;
        dst[(size_t)row * NH + nt * 16 + l15] = (_Float16)acc[nt][r];
      }
  } else {  // V stored transposed: Vt[batch][h][t]
#pragma unroll
    for (int nt = 0; nt < 4; ++nt)
#pragma unroll
      for (int r = 0; r < 8; ++r) {
        const int row   = r0 + r + 8 * hi;
        const int batch = row >> 8;
        const int t     = row & 255;
        Vt[((size_t)batch * NH + nt * 16 + l15) * NT + t] = (_Float16)acc[nt][r];
      }
  }
}

// ---------------------------------------------------------------------------
// Kernel 2: causal flash attention. grid = B*T/16/8 = 512, block = 256.
// One 16-query tile per wave; key loop in 32-wide blocks up to the diagonal.
// K and V fragments are all fetched at the top of the iteration so the long
// softmax VALU section overlaps the V fetch.
// ---------------------------------------------------------------------------
__global__ __launch_bounds__(256) void attn_flash_kernel(
    const _Float16* __restrict__ Qh, const _Float16* __restrict__ Kh,
    const _Float16* __restrict__ Vt, float* __restrict__ out)
{
  __shared__ _Float16 pbuf[8][16 * 32];  // per-wave P staging (C-layout -> A-frag)

  const int wave = threadIdx.x >> 5;
  const int lane = threadIdx.x & 31;
  const int l15  = lane & 15;
  const int hi   = lane >> 4;
  const int tile = blockIdx.x * 8 + wave;     // 16-query tile id
  const int batch = tile >> 4;                // NT/16 = 16 tiles per batch
  const int q0    = (tile & 15) * 16;

  const _Float16* Qb = Qh + (size_t)batch * NT * NH;
  const _Float16* Kb = Kh + (size_t)batch * NT * NH;
  const _Float16* Vb = Vt + (size_t)batch * NH * NT;

  const int abase = hi ? 8 : 0;
  const int kb    = hi ? 16 : 0;
  const float NEG = -__builtin_inff();
  const float scale = 0.125f;                 // 1/sqrt(64)

  // Q A-fragments, two 32-wide K chunks over H=64
  v16h qa[2];
#pragma unroll
  for (int c = 0; c < 2; ++c) {
    const _Float16* qp = Qb + (size_t)(q0 + l15) * NH + c * 32;
    qa[c] = *(const v16h*)(qp + kb);
  }

  v8f o[4] = {};
  float m[8], l[8];
#pragma unroll
  for (int r = 0; r < 8; ++r) { m[r] = NEG; l[r] = 0.0f; }

  for (int s0 = 0; s0 < q0 + 16; s0 += 32) {
    // prefetch next key block's K/V lines (global_prefetch_b8; stays in ws)
    __builtin_prefetch(Kb + (size_t)(s0 + 32 + l15) * NH, 0, 0);
    __builtin_prefetch(Vb + (size_t)l15 * NT + s0 + 32, 0, 0);

    // ---- fetch all fragments for this key block up front (8 x 32B / lane)
    const _Float16* kp = Kb + (size_t)(s0 + l15) * NH + kb;
    v16h kf00 = *(const v16h*)kp;                    // c=0, keys s0..s0+15
    v16h kf01 = *(const v16h*)(kp + 16 * NH);        // c=0, keys s0+16..s0+31
    v16h kf10 = *(const v16h*)(kp + 32);             // c=1
    v16h kf11 = *(const v16h*)(kp + 16 * NH + 32);
    const _Float16* vp = Vb + (size_t)l15 * NT + s0 + kb;
    v16h vf0 = *(const v16h*)vp;                     // h tiles 0..3
    v16h vf1 = *(const v16h*)(vp + 16 * NT);
    v16h vf2 = *(const v16h*)(vp + 32 * NT);
    v16h vf3 = *(const v16h*)(vp + 48 * NT);

    // ---- S = Q K^T for keys [s0, s0+32): two 16x16 f32 tiles
    v8f sA = {}, sB = {};
    sA = WMMA(qa[0], kf00, sA);
    sB = WMMA(qa[0], kf01, sB);
    sA = WMMA(qa[1], kf10, sA);
    sB = WMMA(qa[1], kf11, sB);

    // ---- causal mask + scale + online softmax (rows live in 16-lane halves)
#pragma unroll
    for (int r = 0; r < 8; ++r) {
      const int qi = q0 + r + 8 * hi;
      const int kA = s0 + l15;
      const float vA = (kA      <= qi) ? sA[r] * scale : NEG;
      const float vB = (kA + 16 <= qi) ? sB[r] * scale : NEG;
      float mx = fmaxf(vA, vB);
#pragma unroll
      for (int d = 1; d < 16; d <<= 1) mx = fmaxf(mx, __shfl_xor(mx, d, 32));
      const float mnew = fmaxf(m[r], mx);
      const float corr = __expf(m[r] - mnew);
      const float pA = __expf(vA - mnew);
      const float pB = __expf(vB - mnew);
      float rs = pA + pB;
#pragma unroll
      for (int d = 1; d < 16; d <<= 1) rs += __shfl_xor(rs, d, 32);
      l[r] = l[r] * corr + rs;
      m[r] = mnew;
#pragma unroll
      for (int nt = 0; nt < 4; ++nt) o[nt][r] *= corr;
      pbuf[wave][(r + 8 * hi) * 32 + l15]      = (_Float16)pA;
      pbuf[wave][(r + 8 * hi) * 32 + 16 + l15] = (_Float16)pB;
    }

    // LDS ops from one wave complete in order; wait for stores, then read
    asm volatile("s_wait_dscnt 0" ::: "memory");
    const _Float16* pp = &pbuf[wave][l15 * 32];
    v16h pa = *(const v16h*)(pp + kb);

    // ---- O += P @ V (V fragments already resident)
    o[0] = WMMA(pa, vf0, o[0]);
    o[1] = WMMA(pa, vf1, o[1]);
    o[2] = WMMA(pa, vf2, o[2]);
    o[3] = WMMA(pa, vf3, o[3]);
  }

  // ---- epilogue: out = O / l
#pragma unroll
  for (int nt = 0; nt < 4; ++nt)
#pragma unroll
    for (int r = 0; r < 8; ++r) {
      const int row = batch * NT + q0 + r + 8 * hi;
      out[(size_t)row * NH + nt * 16 + l15] = o[nt][r] / l[r];
    }
}

// ---------------------------------------------------------------------------
extern "C" void kernel_launch(void* const* d_in, const int* in_sizes, int n_in,
                              void* d_out, int out_size, void* d_ws, size_t ws_size,
                              hipStream_t stream) {
  (void)in_sizes; (void)n_in; (void)out_size; (void)ws_size;
  const float* x  = (const float*)d_in[0];
  const float* y  = (const float*)d_in[1];
  const float* Wq = (const float*)d_in[2];
  const float* Wk = (const float*)d_in[3];
  const float* Wv = (const float*)d_in[4];

  const size_t QKV = (size_t)NB * NT * NH;   // 4,194,304 elems
  _Float16* Qh = (_Float16*)d_ws;            // [B*T, H] f16
  _Float16* Kh = Qh + QKV;                   // [B*T, H] f16
  _Float16* Vt = Kh + QKV;                   // [B, H, T] f16 (transposed)
  _Float16* Wt = Vt + QKV;                   // [3][H, C] f16 (transposed wgts)

  attn_wprep_kernel<<<dim3(384), 256, 0, stream>>>(Wq, Wk, Wv, Wt);
  attn_proj_kernel<<<dim3(512, 3), 256, 0, stream>>>(x, y, Wt, Qh, Kh, Vt);
  attn_flash_kernel<<<dim3(512), 256, 0, stream>>>(Qh, Kh, Vt, (float*)d_out);
}
